// KANLinear_83451214561549
// MI455X (gfx1250) — compile-verified, matrix operand order checked by reading, not run on hardware
//
#include <hip/hip_runtime.h>
#include <hip/hip_bf16.h>

// ---------------------------------------------------------------------------
// KAN linear layer for MI455X (gfx1250, wave32, WMMA + Tensor Data Mover).
//   out(1024x512) = [silu(x) | spline_basis(x)] (1024x4608) @ W_packed^T (4608x512)
// Phase 2 GEMM: TDM double-buffered LDS staging + v_wmma_f32_16x16x32_bf16.
// ---------------------------------------------------------------------------

typedef __bf16 bf16_t;
typedef __attribute__((ext_vector_type(16))) __bf16 v16bf;
typedef __attribute__((ext_vector_type(8)))  __bf16 bf16x8;
typedef __attribute__((ext_vector_type(8)))  float  v8f;
typedef __attribute__((ext_vector_type(4)))  unsigned int u32x4;
typedef __attribute__((ext_vector_type(8)))  int i32x8;
typedef __attribute__((ext_vector_type(4)))  int i32x4;

#define B_DIM   1024
#define IN_DIM  512
#define OUT_DIM 512
#define NB      9                    // 1 silu + 8 spline basis
#define KDIM    (IN_DIM * NB)        // 4608, multiple of 64
#define NKNOTS  12                   // GRID_SIZE+1 + 2*SPLINE_ORDER

#define KSLAB     64                 // K elements staged per TDM slab
#define LDS_PITCH 144                // 128B row + 16B TDM pad -> conflict-free
#define A_ROWS    128                // block M tile
#define B_ROWS    64                 // block N tile
#define A_SLAB    (A_ROWS * LDS_PITCH)   // 18432 B
#define B_SLAB    (B_ROWS * LDS_PITCH)   //  9216 B

// -------------------- phase 1a: pack A = [silu | basis0..7] -----------------
__global__ __launch_bounds__(256) void kan_pack_a(
    const float* __restrict__ x,      // (B, IN)
    const float* __restrict__ grid,   // (IN, 12)
    bf16_t* __restrict__ A)           // (B, KDIM) bf16
{
    int idx = blockIdx.x * blockDim.x + threadIdx.x;   // b*IN + i
    if (idx >= B_DIM * IN_DIM) return;
    int i = idx & (IN_DIM - 1);
    float xv = x[idx];

    float t[NKNOTS];
    const float* g = grid + i * NKNOTS;
    #pragma unroll
    for (int j = 0; j < NKNOTS; ++j) t[j] = g[j];

    // Cox-de Boor, order 3: 11 indicators -> 8 basis values
    float v[NKNOTS - 1];
    #pragma unroll
    for (int j = 0; j < NKNOTS - 1; ++j)
        v[j] = (xv >= t[j] && xv < t[j + 1]) ? 1.0f : 0.0f;
    #pragma unroll
    for (int kk = 1; kk <= 3; ++kk) {
        #pragma unroll
        for (int j = 0; j < NKNOTS - 1 - kk; ++j) {
            float left  = (xv - t[j])          / (t[j + kk]     - t[j]);
            float right = (t[j + kk + 1] - xv) / (t[j + kk + 1] - t[j + 1]);
            v[j] = left * v[j] + right * v[j + 1];
        }
    }

    float silu = xv / (1.0f + __expf(-xv));

    bf16_t* dst = A + (size_t)idx * NB;   // == b*KDIM + i*NB
    dst[0] = (bf16_t)silu;
    #pragma unroll
    for (int j = 0; j < 8; ++j) dst[1 + j] = (bf16_t)v[j];
}

// -------------------- phase 1b: pack W (OUT x KDIM, row major) --------------
__global__ __launch_bounds__(256) void kan_pack_w(
    const float* __restrict__ scale_base,    // (OUT, IN)
    const float* __restrict__ scale_spline,  // (OUT, IN)
    const float* __restrict__ spline_weight, // (OUT, IN, 8)
    const float* __restrict__ mask,          // (OUT, IN)
    bf16_t* __restrict__ W)                  // (OUT, KDIM) bf16
{
    int idx = blockIdx.x * blockDim.x + threadIdx.x;   // o*IN + i
    if (idx >= OUT_DIM * IN_DIM) return;

    float m  = mask[idx];
    float sb = scale_base[idx] * m;
    float ss = scale_spline[idx] * m;
    const float* w = spline_weight + (size_t)idx * 8;

    bf16_t* dst = W + (size_t)idx * NB;   // == o*KDIM + i*NB
    dst[0] = (bf16_t)sb;
    #pragma unroll
    for (int j = 0; j < 8; ++j) dst[1 + j] = (bf16_t)(w[j] * ss);
}

// -------------------- TDM 2D tile load: global -> LDS -----------------------
// D# per CDNA5 ISA ch.8: group0 = {count/ctl, lds_addr, global_addr lo,
// global_addr hi | type=2}; group1 = dims/strides; remaining groups zero
// (2D tile). data_size=2B, tile = (KSLAB x rows), pad 16B after every 128B row.
__device__ __forceinline__ unsigned lds_off(const void* p) {
    // flat LDS addresses carry the LDS byte offset in the low 32 bits
    return (unsigned)(size_t)p;
}

__device__ __forceinline__ void tdm_load_2d(const bf16_t* gsrc, unsigned lds_byte,
                                            int tile_rows, int tensor_rows) {
    unsigned long long ga = (unsigned long long)(size_t)gsrc;
    u32x4 g0;
    g0.x = 1u;                                            // count=1, user mode
    g0.y = lds_byte;                                      // LDS dest (bytes)
    g0.z = (unsigned)(ga & 0xFFFFFFFFu);                  // global_addr[31:0]
    g0.w = (unsigned)((ga >> 32) & 0x01FFFFFFu) | (2u << 30); // addr[56:32]|type=2

    i32x8 g1;
    g1[0] = (1 << 16)        // data_size = 2 bytes
          | (1 << 20)        // pad_enable
          | (4 << 22)        // pad_interval code 4 -> every 128 B
          | (3 << 25);       // pad_amount  code 3 -> 4 dwords = 16 B
    g1[1] = (KDIM & 0xFFFF) << 16;            // tensor_dim0[15:0] at bits 63:48
    g1[2] = (tensor_rows & 0xFFFF) << 16;     // dim0 hi=0 | tensor_dim1[15:0]
    g1[3] = (KSLAB & 0xFFFF) << 16;           // dim1 hi=0 | tile_dim0 = 64
    g1[4] = tile_rows & 0xFFFF;               // tile_dim1 | tile_dim2=0
    g1[5] = KDIM;                             // tensor_dim0_stride[31:0]
    g1[6] = 0;                                // stride hi | dim1_stride lo
    g1[7] = 0;

    i32x4 gz4 = {0, 0, 0, 0};
    i32x8 gz8 = {0, 0, 0, 0, 0, 0, 0, 0};
    // amdgpu-toolchain (clang-23) 6-arg form:
    //   (u32x4 g0, i32x8 g1, i32x4, i32x4, i32x8, i32 cpol)
    __builtin_amdgcn_tensor_load_to_lds(g0, g1, gz4, gz4, gz8, 0);
}

// -------------------- LDS fragment load -------------------------------------
// 16-bit A/B fragment (ISA 7.12.2): lane L holds row m = L&15;
// elements 0..7 -> K = (L>>4)*8 + 0..7 ; elements 8..15 -> +16.
__device__ __forceinline__ v16bf load_frag_lds(const unsigned char* p) {
    union { struct { bf16x8 lo, hi; } s; v16bf v; } u;
    u.s.lo = *(const bf16x8*)(p);
    u.s.hi = *(const bf16x8*)(p + 32);   // +16 elements
    return u.v;
}

// -------------------- phase 2: TDM + WMMA GEMM ------------------------------
__global__ __launch_bounds__(256) void kan_gemm(
    const bf16_t* __restrict__ A,   // (B, KDIM)
    const bf16_t* __restrict__ W,   // (OUT, KDIM)
    float* __restrict__ out)        // (B, OUT)
{
    __shared__ __align__(16) unsigned char smem[2][A_SLAB + B_SLAB];

    const int lane  = threadIdx.x & 31;
    const int wave  = threadIdx.x >> 5;        // 0..7
    const int wm    = wave >> 1;               // 0..3 -> 32-row slice of 128
    const int wn    = wave & 1;                // 0..1 -> 32-col slice of 64
    const int m0    = blockIdx.x * A_ROWS;     // block row base
    const int n0    = blockIdx.y * B_ROWS;     // block col base
    const int mrow  = lane & 15;
    const int khalf = (lane >> 4) * 8;

    v8f acc00 = {}, acc01 = {}, acc10 = {}, acc11 = {};

    // prologue: stage slab 0 into buffer 0 (wave0: A, wave1: W)
    if (wave == 0)
        tdm_load_2d(A + (size_t)m0 * KDIM, lds_off(&smem[0][0]), A_ROWS, B_DIM);
    else if (wave == 1)
        tdm_load_2d(W + (size_t)n0 * KDIM, lds_off(&smem[0][A_SLAB]), B_ROWS, OUT_DIM);

    const int NS = KDIM / KSLAB;   // 72 slabs
    for (int ks = 0; ks < NS; ++ks) {
        const int  buf  = ks & 1;
        const bool more = (ks + 1) < NS;

        if (more) {   // kick next slab into the other buffer
            const int k1 = (ks + 1) * KSLAB;
            if (wave == 0)
                tdm_load_2d(A + (size_t)m0 * KDIM + k1,
                            lds_off(&smem[buf ^ 1][0]), A_ROWS, B_DIM);
            else if (wave == 1)
                tdm_load_2d(W + (size_t)n0 * KDIM + k1,
                            lds_off(&smem[buf ^ 1][A_SLAB]), B_ROWS, OUT_DIM);
        }
        if (wave < 2) {   // issuing waves wait for the current slab only
            if (more) __builtin_amdgcn_s_wait_tensorcnt(1);
            else      __builtin_amdgcn_s_wait_tensorcnt(0);
        }
        __syncthreads();   // slab `ks` visible to all waves

        const unsigned char* sa = &smem[buf][0];
        const unsigned char* sb = &smem[buf][A_SLAB];
        #pragma unroll
        for (int kk = 0; kk < KSLAB; kk += 32) {
            const int kb = 2 * (kk + khalf);
            v16bf af0 = load_frag_lds(sa + (wm * 32      + mrow) * LDS_PITCH + kb);
            v16bf af1 = load_frag_lds(sa + (wm * 32 + 16 + mrow) * LDS_PITCH + kb);
            v16bf bf0 = load_frag_lds(sb + (wn * 32      + mrow) * LDS_PITCH + kb);
            v16bf bf1 = load_frag_lds(sb + (wn * 32 + 16 + mrow) * LDS_PITCH + kb);

            acc00 = __builtin_amdgcn_wmma_f32_16x16x32_bf16(false, af0, false, bf0,
                                                            (short)0, acc00, false, false);
            acc01 = __builtin_amdgcn_wmma_f32_16x16x32_bf16(false, af0, false, bf1,
                                                            (short)0, acc01, false, false);
            acc10 = __builtin_amdgcn_wmma_f32_16x16x32_bf16(false, af1, false, bf0,
                                                            (short)0, acc10, false, false);
            acc11 = __builtin_amdgcn_wmma_f32_16x16x32_bf16(false, af1, false, bf1,
                                                            (short)0, acc11, false, false);
        }
        __syncthreads();   // done reading `buf` before it is refilled
    }

    // C/D layout (ISA 7.12.2): VGPR r, lanes 0-15 -> M=r, N=lane;
    //                                   lanes 16-31 -> M=r+8, N=lane-16.
    const int mw   = m0 + wm * 32;
    const int nw   = n0 + wn * 32;
    const int orow = (lane >> 4) * 8;
    const int ocol = lane & 15;
    #pragma unroll
    for (int r = 0; r < 8; ++r) {
        out[(size_t)(mw      + orow + r) * OUT_DIM + (nw      + ocol)] = acc00[r];
        out[(size_t)(mw      + orow + r) * OUT_DIM + (nw + 16 + ocol)] = acc01[r];
        out[(size_t)(mw + 16 + orow + r) * OUT_DIM + (nw      + ocol)] = acc10[r];
        out[(size_t)(mw + 16 + orow + r) * OUT_DIM + (nw + 16 + ocol)] = acc11[r];
    }
}

// ---------------------------------------------------------------------------
extern "C" void kernel_launch(void* const* d_in, const int* in_sizes, int n_in,
                              void* d_out, int out_size, void* d_ws, size_t ws_size,
                              hipStream_t stream) {
    const float* x             = (const float*)d_in[0];
    const float* scale_base    = (const float*)d_in[1];
    const float* scale_spline  = (const float*)d_in[2];
    const float* spline_weight = (const float*)d_in[3];
    const float* grid          = (const float*)d_in[4];
    const float* mask          = (const float*)d_in[5];
    float*       out           = (float*)d_out;

    // workspace: A_packed (B x KDIM bf16), then W_packed (OUT x KDIM bf16)
    bf16_t* A_packed = (bf16_t*)d_ws;
    bf16_t* W_packed = (bf16_t*)((char*)d_ws + (size_t)B_DIM * KDIM * sizeof(bf16_t));

    kan_pack_a<<<(B_DIM * IN_DIM + 255) / 256, 256, 0, stream>>>(x, grid, A_packed);
    kan_pack_w<<<(OUT_DIM * IN_DIM + 255) / 256, 256, 0, stream>>>(
        scale_base, scale_spline, spline_weight, mask, W_packed);

    dim3 ggrid(B_DIM / A_ROWS, OUT_DIM / B_ROWS);   // 8 x 8 blocks, 8 waves each
    kan_gemm<<<ggrid, 256, 0, stream>>>(A_packed, W_packed, out);
}
